// MoLAAdapter_85761906967163
// MI455X (gfx1250) — compile-verified, matrix operand order checked by reading, not run on hardware
//
#include <hip/hip_runtime.h>
#include <hip/hip_bf16.h>

// ---------------------------------------------------------------------------
// MoLA adapter for MI455X (gfx1250, wave32, WMMA).
// Split-precision bf16 (hi+lo) WMMA with f32 accumulation:
//   x*w ~= xh*wh + xh*wl + xl*wh   (fp32-class accuracy, bf16 WMMA rate)
// All GEMM operands are pre-packed ONCE into WMMA fragment-major bf16 panels
// ([tile16][kchunk32][lane][16]), so the hot loops are pure
// global_load_b128 -> v_wmma with no per-iteration conversion VALU.
// Pipeline: pack(x,W,PB) -> k_hl (router+LoRA-h) -> k_gate -> k_main
// (base GEMM + fused bias + LoRA epilogue; d_out written exactly once).
// ---------------------------------------------------------------------------

typedef __attribute__((ext_vector_type(16))) __bf16 v16bf;
typedef __attribute__((ext_vector_type(8)))  float  v8f;

#define D_IN   2048
#define D_OUT  2048
#define NTOK   8192   // B*S
#define NEXP   8
#define RANK   8
#define LSCALE 2.0f   // ALPHA / R
#define PBROWS 80     // 8 router + 64 (E*R) + 8 pad
#define PBTILES 5
#define TOKSTRIDE 20  // dwords per token record

#define KC     64     // D_IN / 32 k-chunks
#define FRAG   16     // bf16 per lane per fragment
#define MT_X   512    // NTOK/16 row tiles of x
#define NT_W   128    // D_OUT/16 col tiles of W

#define TM 128
#define TN 128

__device__ __forceinline__ void split_bf16(float x, __bf16& h, __bf16& l) {
    h = (__bf16)x;                 // RNE to bf16 (hi)
    l = (__bf16)(x - (float)h);    // residual (lo)
}

// fragment element index inside a panel: ((tile*KC + kc)*32 + lane)*FRAG
__device__ __forceinline__ size_t frag_off(int tile, int kc, int lane) {
    return (((size_t)tile * KC + kc) * 32 + lane) * FRAG;
}

// ---------------------------------------------------------------------------
// Pack x [8192x2048] f32 -> fragment-major bf16 hi/lo panels.
// One thread builds one lane-slice (16 bf16) of one fragment.
// A-layout: lanes 0-15 row=m, K {0..7,16..23}; lanes 16-31 K {8..15,24..31}.
// ---------------------------------------------------------------------------
__global__ __launch_bounds__(256) void k_packx(const float* __restrict__ x,
                                               __bf16* __restrict__ XPh,
                                               __bf16* __restrict__ XPl) {
    int t = blockIdx.x * 256 + threadIdx.x;        // 0 .. MT_X*KC*32-1
    int lane = t & 31;
    int kc   = (t >> 5) & (KC - 1);
    int mt   = t >> 11;
    int row  = mt * 16 + (lane & 15);
    int kb   = (lane >> 4) << 3;
    const float* src = x + (size_t)row * D_IN + kc * 32;
    v16bf h, l;
#pragma unroll
    for (int j = 0; j < 8; ++j) {
        __bf16 h0, l0, h1, l1;
        split_bf16(src[kb + j],      h0, l0);
        split_bf16(src[kb + 16 + j], h1, l1);
        h[j] = h0; l[j] = l0; h[8 + j] = h1; l[8 + j] = l1;
    }
    size_t off = (size_t)t * FRAG;
    *(v16bf*)(XPh + off) = h;
    *(v16bf*)(XPl + off) = l;
}

// Same for W_base [2048x2048] (B-matrix panel: tile = 16 output features).
__global__ __launch_bounds__(256) void k_packw(const float* __restrict__ Wb,
                                               __bf16* __restrict__ WPh,
                                               __bf16* __restrict__ WPl) {
    int t = blockIdx.x * 256 + threadIdx.x;        // 0 .. NT_W*KC*32-1
    int lane = t & 31;
    int kc   = (t >> 5) & (KC - 1);
    int nt   = t >> 11;
    int row  = nt * 16 + (lane & 15);
    int kb   = (lane >> 4) << 3;
    const float* src = Wb + (size_t)row * D_IN + kc * 32;
    v16bf h, l;
#pragma unroll
    for (int j = 0; j < 8; ++j) {
        __bf16 h0, l0, h1, l1;
        split_bf16(src[kb + j],      h0, l0);
        split_bf16(src[kb + 16 + j], h1, l1);
        h[j] = h0; l[j] = l0; h[8 + j] = h1; l[8 + j] = l1;
    }
    size_t off = (size_t)t * FRAG;
    *(v16bf*)(WPh + off) = h;
    *(v16bf*)(WPl + off) = l;
}

// Pack [W_router ; A_w ; zeros] (80 logical rows) into 5 fragment tiles.
__global__ __launch_bounds__(256) void k_packpb(const float* __restrict__ Wr,
                                                const float* __restrict__ Aw,
                                                __bf16* __restrict__ PBh,
                                                __bf16* __restrict__ PBl) {
    int t = blockIdx.x * 256 + threadIdx.x;        // 0 .. PBTILES*KC*32-1
    if (t >= PBTILES * KC * 32) return;
    int lane = t & 31;
    int kc   = (t >> 5) & (KC - 1);
    int nt   = t >> 11;
    int row  = nt * 16 + (lane & 15);              // 0..79
    int kb   = (lane >> 4) << 3;
    const float* src = (row < NEXP) ? (Wr + (size_t)row * D_IN)
                     : (row < NEXP + NEXP * RANK) ? (Aw + (size_t)(row - NEXP) * D_IN)
                     : nullptr;
    v16bf h, l;
#pragma unroll
    for (int j = 0; j < 8; ++j) {
        float f0 = src ? src[kc * 32 + kb + j]      : 0.0f;
        float f1 = src ? src[kc * 32 + kb + 16 + j] : 0.0f;
        __bf16 h0, l0, h1, l1;
        split_bf16(f0, h0, l0);
        split_bf16(f1, h1, l1);
        h[j] = h0; l[j] = l0; h[8 + j] = h1; l[8 + j] = l1;
    }
    size_t off = (size_t)t * FRAG;
    *(v16bf*)(PBh + off) = h;
    *(v16bf*)(PBl + off) = l;
}

// ---------------------------------------------------------------------------
// hl[tok][0..79] = x @ [router;A]^T. One wave per 16-token tile; pure
// fragment loads + split-bf16 WMMA (15 WMMA per k-chunk).
// ---------------------------------------------------------------------------
__global__ __launch_bounds__(256) void k_hl(const __bf16* __restrict__ XPh,
                                            const __bf16* __restrict__ XPl,
                                            const __bf16* __restrict__ PBh,
                                            const __bf16* __restrict__ PBl,
                                            float* __restrict__ hl) {
    const int lane = threadIdx.x & 31;
    const int w    = threadIdx.x >> 5;
    const int mt   = blockIdx.x * 8 + w;           // row tile 0..511
    const int row16 = lane & 15;
    const int mhi   = (lane >> 4) * 8;

    v8f acc[PBTILES];
#pragma unroll
    for (int t = 0; t < PBTILES; ++t)
#pragma unroll
        for (int j = 0; j < 8; ++j) acc[t][j] = 0.0f;

    for (int kc = 0; kc < KC; ++kc) {
        size_t ao = frag_off(mt, kc, lane);
        v16bf ah = *(const v16bf*)(XPh + ao);
        v16bf al = *(const v16bf*)(XPl + ao);
#pragma unroll
        for (int t = 0; t < PBTILES; ++t) {
            size_t bo = frag_off(t, kc, lane);
            v16bf bh = *(const v16bf*)(PBh + bo);
            v16bf bl = *(const v16bf*)(PBl + bo);
            acc[t] = __builtin_amdgcn_wmma_f32_16x16x32_bf16(false, ah, false, bh, (short)0, acc[t], false, false);
            acc[t] = __builtin_amdgcn_wmma_f32_16x16x32_bf16(false, ah, false, bl, (short)0, acc[t], false, false);
            acc[t] = __builtin_amdgcn_wmma_f32_16x16x32_bf16(false, al, false, bh, (short)0, acc[t], false, false);
        }
    }
#pragma unroll
    for (int t = 0; t < PBTILES; ++t)
#pragma unroll
        for (int i = 0; i < 8; ++i) {
            int tok = mt * 16 + i + mhi;
            hl[(size_t)tok * PBROWS + t * 16 + row16] = acc[t][i];
        }
}

// ---------------------------------------------------------------------------
// Per-token top-2 + softmax; fold gate*SCALE into the selected h rows.
// Record: [e0(int), e1(int), h0*g0*S (8), h1*g1*S (8), pad, pad]
// ---------------------------------------------------------------------------
__global__ __launch_bounds__(256) void k_gate(const float* __restrict__ hl,
                                              float* __restrict__ tokbuf) {
    int tok = blockIdx.x * 256 + threadIdx.x;
    if (tok >= NTOK) return;
    const float* row = hl + (size_t)tok * PBROWS;

    int e0 = 0; float v0 = row[0];
#pragma unroll
    for (int e = 1; e < NEXP; ++e) { float v = row[e]; if (v > v0) { v0 = v; e0 = e; } }
    int e1 = (e0 == 0) ? 1 : 0; float v1 = row[e1];
#pragma unroll
    for (int e = 0; e < NEXP; ++e) {
        if (e == e0) continue;
        float v = row[e];
        if (v > v1) { v1 = v; e1 = e; }
    }
    float x1  = __expf(v1 - v0);
    float inv = 1.0f / (1.0f + x1);
    float g0 = inv * LSCALE;
    float g1 = x1 * inv * LSCALE;

    float* o = tokbuf + (size_t)tok * TOKSTRIDE;
    o[0] = __int_as_float(e0);
    o[1] = __int_as_float(e1);
#pragma unroll
    for (int r = 0; r < RANK; ++r) {
        o[2 + r]  = row[NEXP + e0 * RANK + r] * g0;
        o[10 + r] = row[NEXP + e1 * RANK + r] * g1;
    }
    o[18] = 0.0f; o[19] = 0.0f;
}

// ---------------------------------------------------------------------------
// Base GEMM y = x @ W^T (+bias) with fused LoRA epilogue.
// 128x128 tile / block, 8 waves (4x2), operands from fragment-major panels:
// inner loop = 24 global_load_b128 + 24 v_wmma per 32-wide K chunk.
// Gating records staged to LDS with async global->LDS loads (ASYNCcnt).
// ---------------------------------------------------------------------------
__global__ __launch_bounds__(256) void k_main(const __bf16* __restrict__ XPh,
                                              const __bf16* __restrict__ XPl,
                                              const __bf16* __restrict__ WPh,
                                              const __bf16* __restrict__ WPl,
                                              const float* __restrict__ bb,
                                              const float* __restrict__ Bw,
                                              const float* __restrict__ tokbuf,
                                              float* __restrict__ out) {
    __shared__ float ltok[TM][TOKSTRIDE];          // 2560 dwords

    const int tid  = threadIdx.x;
    const int lane = tid & 31;
    const int w    = tid >> 5;
    const int bm   = blockIdx.y * TM;
    const int bn   = blockIdx.x * TN;

    // Stage gating records into LDS via CDNA5 async global->LDS loads.
    {
        unsigned lds0 = (unsigned)(uintptr_t)(&ltok[0][0]);
        const float* gsrc = tokbuf + (size_t)bm * TOKSTRIDE;
        for (int i = tid; i < TM * TOKSTRIDE; i += 256) {
            unsigned laddr = lds0 + (unsigned)(i * 4);
            unsigned long long gaddr = (unsigned long long)(uintptr_t)(gsrc + i);
            asm volatile("global_load_async_to_lds_b32 %0, %1, off"
                         :: "v"(laddr), "v"(gaddr) : "memory");
        }
        asm volatile("s_wait_asynccnt 0x0" ::: "memory");
    }
    __syncthreads();

    const int wm = w >> 1;                         // 0..3 -> 32-row slab
    const int wn = w & 1;                          // 0..1 -> 64-col slab
    const int row16 = lane & 15;
    const int mhi   = (lane >> 4) * 8;

    v8f acc[2][4];
#pragma unroll
    for (int a = 0; a < 2; ++a)
#pragma unroll
        for (int b = 0; b < 4; ++b)
#pragma unroll
            for (int j = 0; j < 8; ++j) acc[a][b][j] = 0.0f;

    const int mt0 = (bm >> 4) + wm * 2;            // first of 2 row tiles
    const int nt0 = (bn >> 4) + wn * 4;            // first of 4 col tiles

    for (int kc = 0; kc < KC; ++kc) {
        v16bf ah[2], al[2];
#pragma unroll
        for (int a = 0; a < 2; ++a) {
            size_t ao = frag_off(mt0 + a, kc, lane);
            ah[a] = *(const v16bf*)(XPh + ao);
            al[a] = *(const v16bf*)(XPl + ao);
        }
#pragma unroll
        for (int b = 0; b < 4; ++b) {
            size_t bo = frag_off(nt0 + b, kc, lane);
            v16bf bh = *(const v16bf*)(WPh + bo);
            v16bf bl = *(const v16bf*)(WPl + bo);
#pragma unroll
            for (int a = 0; a < 2; ++a) {
                acc[a][b] = __builtin_amdgcn_wmma_f32_16x16x32_bf16(false, ah[a], false, bh, (short)0, acc[a][b], false, false);
                acc[a][b] = __builtin_amdgcn_wmma_f32_16x16x32_bf16(false, ah[a], false, bl, (short)0, acc[a][b], false, false);
                acc[a][b] = __builtin_amdgcn_wmma_f32_16x16x32_bf16(false, al[a], false, bh, (short)0, acc[a][b], false, false);
            }
        }
        if (kc + 1 < KC) {  // pull next chunk toward GL2 while WMMAs run
            __builtin_prefetch(XPh + frag_off(mt0, kc + 1, lane), 0, 1);
            __builtin_prefetch(WPh + frag_off(nt0, kc + 1, lane), 0, 1);
        }
    }

    // Fused epilogue: + bias + sum_k gate_k*SCALE * (h_k . B_w[e_k][n][:])
#pragma unroll
    for (int a = 0; a < 2; ++a) {
#pragma unroll
        for (int i = 0; i < 8; ++i) {
            const int mloc = wm * 32 + a * 16 + i + mhi;
            const float* tk = &ltok[mloc][0];
            const int e0 = __float_as_int(tk[0]);
            const int e1 = __float_as_int(tk[1]);
#pragma unroll
            for (int b = 0; b < 4; ++b) {
                const int n = bn + wn * 64 + b * 16 + row16;
                const float* bw0 = Bw + ((size_t)e0 * D_OUT + n) * RANK;
                const float* bw1 = Bw + ((size_t)e1 * D_OUT + n) * RANK;
                float f = 0.0f;
#pragma unroll
                for (int r = 0; r < RANK; ++r)
                    f += tk[2 + r] * bw0[r] + tk[10 + r] * bw1[r];
                out[(size_t)(bm + mloc) * D_OUT + n] = acc[a][b][i] + bb[n] + f;
            }
        }
    }
}

// ---------------------------------------------------------------------------
// Launch
// ---------------------------------------------------------------------------
extern "C" void kernel_launch(void* const* d_in, const int* in_sizes, int n_in,
                              void* d_out, int out_size, void* d_ws, size_t ws_size,
                              hipStream_t stream) {
    const float* x  = (const float*)d_in[0];   // [4,2048,2048]
    const float* Wb = (const float*)d_in[1];   // [2048,2048]
    const float* bb = (const float*)d_in[2];   // [2048]
    const float* Wr = (const float*)d_in[3];   // [8,2048]
    const float* Aw = (const float*)d_in[4];   // [8,8,2048]
    const float* Bw = (const float*)d_in[5];   // [8,2048,8]
    float* out = (float*)d_out;

    char* ws = (char*)d_ws;
    // panel sizes (bytes): XP 2x33554432, WP 2x8388608, PB 2x327680
    __bf16* XPh = (__bf16*)(ws);
    __bf16* XPl = (__bf16*)(ws + 33554432ull);
    __bf16* WPh = (__bf16*)(ws + 67108864ull);
    __bf16* WPl = (__bf16*)(ws + 75497472ull);
    __bf16* PBh = (__bf16*)(ws + 83886080ull);
    __bf16* PBl = (__bf16*)(ws + 84213760ull);
    float*  hl  = (float*)(ws + 84541440ull);  // 8192*80*4
    float*  tok = (float*)(ws + 87162880ull);  // 8192*20*4

    hipLaunchKernelGGL(k_packx,  dim3(MT_X * KC * 32 / 256), dim3(256), 0, stream, x,  XPh, XPl);
    hipLaunchKernelGGL(k_packw,  dim3(NT_W * KC * 32 / 256), dim3(256), 0, stream, Wb, WPh, WPl);
    hipLaunchKernelGGL(k_packpb, dim3((PBTILES * KC * 32 + 255) / 256), dim3(256), 0, stream, Wr, Aw, PBh, PBl);
    hipLaunchKernelGGL(k_hl,     dim3(MT_X / 8), dim3(256), 0, stream, XPh, XPl, PBh, PBl, hl);
    hipLaunchKernelGGL(k_gate,   dim3(NTOK / 256), dim3(256), 0, stream, hl, tok);
    hipLaunchKernelGGL(k_main,   dim3(D_OUT / TN, NTOK / TM), dim3(256), 0, stream,
                       XPh, XPl, WPh, WPl, bb, Bw, tok, out);
}
